// MultiHeadAttention_44770739093872
// MI455X (gfx1250) — matrix operand order repaired and checked by reading, not compiled, not run on hardware
//
#include <hip/hip_runtime.h>
#include <hip/hip_bf16.h>

// ---------------- problem constants ----------------
#define B_  2
#define T_  2048
#define D_  1024
#define H_  16
#define DK_ 64
#define M_  (B_ * T_)   // 4096 rows

typedef __attribute__((ext_vector_type(16))) __bf16 v16bf;
typedef __attribute__((ext_vector_type(8)))  float  v8f;

// ---------------- TDM (Tensor Data Mover) support, probe-verified signatures ----------------
#if defined(__gfx1250__) && __has_builtin(__builtin_amdgcn_tensor_load_to_lds) && \
    __has_builtin(__builtin_amdgcn_s_wait_tensorcnt)
#define HAVE_TDM 1
#else
#define HAVE_TDM 0
#endif

#if HAVE_TDM
typedef unsigned v4u_ __attribute__((ext_vector_type(4)));
typedef int      v8i_ __attribute__((ext_vector_type(8)));
typedef int      v4i_ __attribute__((ext_vector_type(4)));

// 2-D tile DMA: global (row-major, 2-byte elems, row stride stride0 elems) -> LDS,
// with LDS padding of 4 DWORDs (16B) after every 32 DWORDs (128B = 64 elems),
// i.e. destination rows have a 144B stride == our padded [..][72] LDS arrays.
__device__ __forceinline__ void tdm_load_2d(const void* gptr, unsigned lds_byte_addr,
                                            unsigned tensor_d0, unsigned tensor_d1,
                                            unsigned stride0,
                                            unsigned tile_d0, unsigned tile_d1) {
  unsigned long long ga = (unsigned long long)(size_t)gptr;
  v4u_ g0;
  g0[0] = 1u;                                              // count=1, user mode
  g0[1] = lds_byte_addr;                                   // LDS dest (bytes)
  g0[2] = (unsigned)(ga & 0xFFFFFFFFu);                    // global_addr[31:0]
  g0[3] = (unsigned)((ga >> 32) & 0x01FFFFFFu) | (2u << 30); // addr[56:32] | type=2
  v8i_ g1;
  g1[0] = (int)((1u << 16)      // data_size = 1 -> 2 bytes
              | (1u << 20)      // pad_enable
              | (4u << 22)      // pad_interval: 32 DWORDs
              | (3u << 25));    // pad_amount:   4 DWORDs
  g1[1] = (int)((tensor_d0 & 0xFFFFu) << 16);              // dim0[15:0]
  g1[2] = (int)((tensor_d0 >> 16) | ((tensor_d1 & 0xFFFFu) << 16));
  g1[3] = (int)((tensor_d1 >> 16) | (tile_d0 << 16));      // dim1[31:16] | tile_dim0
  g1[4] = (int)(tile_d1 & 0xFFFFu);                        // tile_dim1 (tile_dim2=0)
  g1[5] = (int)stride0;                                    // dim0_stride[31:0]
  g1[6] = 0;
  g1[7] = 0;
  v4i_ zz = {0, 0, 0, 0};
#if __clang_major__ >= 23
  v8i_ zz8 = {0, 0, 0, 0, 0, 0, 0, 0};
  __builtin_amdgcn_tensor_load_to_lds(g0, g1, zz, zz, zz8, 0);
#else
  __builtin_amdgcn_tensor_load_to_lds(g0, g1, zz, zz, 0);
#endif
}
#endif  // HAVE_TDM

// ---------------- bf16 helpers (bit-exact RNE) ----------------
__device__ __forceinline__ __bf16 f2bf(float f) {
  unsigned u = __builtin_bit_cast(unsigned, f);
  unsigned r = u + 0x7FFFu + ((u >> 16) & 1u);
  unsigned short h = (unsigned short)(r >> 16);
  return __builtin_bit_cast(__bf16, h);
}

// ---------------- WMMA wrapper ----------------
__device__ __forceinline__ v8f wmma_bf16(v16bf a, v16bf b, v8f c) {
  return __builtin_amdgcn_wmma_f32_16x16x32_bf16(
      false, a, false, b, (short)0, c, false, false);
}

// A fragment (16x32 bf16): lanes 0-15 -> M=lane, K=0..7,16..23; lanes 16-31 -> M=lane-16, K=8..15,24..31.
__device__ __forceinline__ v16bf load_a_frag(const __bf16* base, int ld) {
  int lane = threadIdx.x & 31;
  int row  = lane & 15;
  int k0   = (lane < 16) ? 0 : 8;
  const __bf16* p = base + (size_t)row * ld;
  v16bf a;
#pragma unroll
  for (int i = 0; i < 8; ++i) a[i] = p[k0 + i];
#pragma unroll
  for (int i = 0; i < 8; ++i) a[8 + i] = p[k0 + 16 + i];
  return a;
}

// B fragment (32x16 bf16) from N-major storage: lanes 0-15 -> N=lane, K=0..7,16..23; lanes 16-31 -> K=8..15,24..31.
__device__ __forceinline__ v16bf load_b_frag(const __bf16* base, int ld) {
  int lane = threadIdx.x & 31;
  int col  = lane & 15;
  int k0   = (lane < 16) ? 0 : 8;
  const __bf16* p = base + (size_t)col * ld + k0;
  v16bf b;
#pragma unroll
  for (int i = 0; i < 8; ++i) b[i] = p[i];
#pragma unroll
  for (int i = 0; i < 8; ++i) b[8 + i] = p[16 + i];
  return b;
}

// ---------------- stage 1: fp32 -> bf16 cast (vectorized) ----------------
__global__ void cast_f32_to_bf16_v4(const float4* __restrict__ src,
                                    __bf16* __restrict__ dst, int n4) {
  int i = blockIdx.x * blockDim.x + threadIdx.x;
  int stride = gridDim.x * blockDim.x;
  for (; i < n4; i += stride) {
    float4 f = src[i];
    __align__(8) __bf16 q[4] = { f2bf(f.x), f2bf(f.y), f2bf(f.z), f2bf(f.w) };
    *(uint2*)&dst[(size_t)i * 4] = *(uint2*)q;
  }
}

// ---------------- epilogue store overloads ----------------
__device__ __forceinline__ void store_out(float* p, float v)  { *p = v; }
__device__ __forceinline__ void store_out(__bf16* p, float v) { *p = f2bf(v); }

// ---------------- stages 2 & 4: bf16 WMMA GEMM, C = A(MxK)*B(KxN) + bias ----------------
// 256 threads = 8 waves; block tile 128x128; BK=64; wave tile 32x64 -> 16 WMMAs/iter/wave.
// A tile arrives via TDM DMA (wave 0 issues) while all threads stage the transposed B tile.
template <typename OutT>
__global__ void __launch_bounds__(256)
gemm_bf16_wmma(const __bf16* __restrict__ A, const __bf16* __restrict__ Bm,
               const float* __restrict__ bias, OutT* __restrict__ C,
               int M, int N, int K) {
  __shared__ __bf16 As[128][72];   // 144B rows (TDM pad config matches)
  __shared__ __bf16 Bt[128][72];   // B tile transposed: Bt[n][k]

  const int tid  = threadIdx.x;
  const int lane = tid & 31;
  const int wave = tid >> 5;
  const int wm   = wave >> 1;      // 0..3
  const int wn   = wave & 1;       // 0..1
  const int m0   = blockIdx.x * 128;
  const int n0   = blockIdx.y * 128;
#if HAVE_TDM
  const unsigned ldsAs = (unsigned)(size_t)&As[0][0];
#endif

  v8f zero = {};
  v8f acc[2][4];
#pragma unroll
  for (int i = 0; i < 2; ++i)
#pragma unroll
    for (int j = 0; j < 4; ++j) acc[i][j] = zero;

  for (int kk = 0; kk < K; kk += 64) {
#if HAVE_TDM
    // --- A tile 128x64 via Tensor Data Mover (one issue per workgroup) ---
    if (wave == 0) {
      tdm_load_2d(A + (size_t)m0 * K + kk, ldsAs,
                  (unsigned)K, (unsigned)M, (unsigned)K, 64u, 128u);
    }
#else
    {
      int row = tid >> 1;
      int seg = (tid & 1) * 32;
      const uint4* g = (const uint4*)(A + (size_t)(m0 + row) * K + kk + seg);
      uint4 d0 = g[0], d1 = g[1], d2 = g[2], d3 = g[3];
      *(uint4*)&As[row][seg]      = d0;
      *(uint4*)&As[row][seg + 8]  = d1;
      *(uint4*)&As[row][seg + 16] = d2;
      *(uint4*)&As[row][seg + 24] = d3;
    }
#endif
    // --- B tile 64x128 -> transposed Bt[n][k]; 4x8 sub-blocks, b64 LDS stores ---
    {
      int k0b = (tid >> 4) * 4;
      int nsb = (tid & 15) * 8;
      const __bf16* g = Bm + (size_t)(kk + k0b) * N + n0 + nsb;
      uint4 r0 = *(const uint4*)(g + 0 * N);
      uint4 r1 = *(const uint4*)(g + 1 * N);
      uint4 r2 = *(const uint4*)(g + 2 * N);
      uint4 r3 = *(const uint4*)(g + 3 * N);
      if (kk + 64 < K)
        __builtin_prefetch(Bm + (size_t)(kk + 64 + k0b) * N + n0 + nsb, 0, 1);
      __bf16 t0[8], t1[8], t2[8], t3[8];
      *(uint4*)t0 = r0; *(uint4*)t1 = r1; *(uint4*)t2 = r2; *(uint4*)t3 = r3;
#pragma unroll
      for (int j = 0; j < 8; ++j) {
        __align__(8) __bf16 q[4] = { t0[j], t1[j], t2[j], t3[j] };
        *(uint2*)&Bt[nsb + j][k0b] = *(uint2*)q;
      }
    }
#if HAVE_TDM
    if (wave == 0) __builtin_amdgcn_s_wait_tensorcnt(0);
#endif
    __syncthreads();

#pragma unroll
    for (int kc = 0; kc < 2; ++kc) {
      const int ko = kc * 32;
      v16bf a0 = load_a_frag(&As[wm * 32][ko],      72);
      v16bf a1 = load_a_frag(&As[wm * 32 + 16][ko], 72);
      v16bf b0 = load_b_frag(&Bt[wn * 64][ko],      72);
      v16bf b1 = load_b_frag(&Bt[wn * 64 + 16][ko], 72);
      v16bf b2 = load_b_frag(&Bt[wn * 64 + 32][ko], 72);
      v16bf b3 = load_b_frag(&Bt[wn * 64 + 48][ko], 72);
      acc[0][0] = wmma_bf16(a0, b0, acc[0][0]);
      acc[1][0] = wmma_bf16(a1, b0, acc[1][0]);
      acc[0][1] = wmma_bf16(a0, b1, acc[0][1]);
      acc[1][1] = wmma_bf16(a1, b1, acc[1][1]);
      acc[0][2] = wmma_bf16(a0, b2, acc[0][2]);
      acc[1][2] = wmma_bf16(a1, b2, acc[1][2]);
      acc[0][3] = wmma_bf16(a0, b3, acc[0][3]);
      acc[1][3] = wmma_bf16(a1, b3, acc[1][3]);
    }
    __syncthreads();
  }

  const int colL = lane & 15;
  const int rowB = (lane < 16) ? 0 : 8;
#pragma unroll
  for (int i2 = 0; i2 < 2; ++i2) {
#pragma unroll
    for (int j2 = 0; j2 < 4; ++j2) {
      int n = n0 + wn * 64 + j2 * 16 + colL;
      float bv = bias[n];
#pragma unroll
      for (int e = 0; e < 8; ++e) {
        int m = m0 + wm * 32 + i2 * 16 + rowB + e;
        store_out(&C[(size_t)m * N + n], acc[i2][j2][e] + bv);
      }
    }
  }
}

// ---------------- stage 3: causal flash attention ----------------
// Grid: (T/128, H, B). 256 threads = 8 waves; wave owns 16 query rows.
// K tile DMA'd by TDM (wave 0); V tile transposed by all threads in parallel.
__global__ void __launch_bounds__(256)
attn_kernel(const __bf16* __restrict__ Q, const __bf16* __restrict__ Km,
            const __bf16* __restrict__ V, __bf16* __restrict__ O) {
  __shared__ __bf16 Ks[64][72];          // Ks[key][dk], 144B rows (TDM pad matches)
  __shared__ __bf16 Vt[64][72];          // Vt[dk][key]
  __shared__ __bf16 Ps[8][16][40];       // per-wave P scratch (16 q x 32 keys)

  const int tid   = threadIdx.x;
  const int lane  = tid & 31;
  const int wave  = tid >> 5;
  const int qbase = blockIdx.x * 128;
  const int h     = blockIdx.y;
  const int b     = blockIdx.z;
  const int q0    = qbase + wave * 16;
  const size_t rowOff = (size_t)b * T_;
  const int hcol  = h * DK_;
#if HAVE_TDM
  const unsigned ldsKs = (unsigned)(size_t)&Ks[0][0];
#endif

  v16bf aq0 = load_a_frag(Q + (rowOff + q0) * D_ + hcol,      D_);
  v16bf aq1 = load_a_frag(Q + (rowOff + q0) * D_ + hcol + 32, D_);

  v8f zero = {};
  v8f accO[4];
#pragma unroll
  for (int g = 0; g < 4; ++g) accO[g] = zero;
  float mrow[8], lrow[8];
#pragma unroll
  for (int e = 0; e < 8; ++e) { mrow[e] = -1e30f; lrow[e] = 0.0f; }

  const int colL = lane & 15;
  const int rowB = (lane < 16) ? 0 : 8;

  const int nTiles = (qbase + 128) / 64;   // causal: keys 0 .. qbase+127
  for (int t = 0; t < nTiles; ++t) {
    const int kt0 = t * 64;
#if HAVE_TDM
    if (wave == 0) {
      tdm_load_2d(Km + (rowOff + kt0) * D_ + hcol, ldsKs,
                  (unsigned)D_, (unsigned)M_, (unsigned)D_, (unsigned)DK_, 64u);
    }
#else
    {
      int key = tid >> 2;
      int seg = (tid & 3) * 16;
      const uint4* gk = (const uint4*)(Km + (rowOff + kt0 + key) * D_ + hcol + seg);
      uint4 d0 = gk[0];
      uint4 d1 = gk[1];
      *(uint4*)&Ks[key][seg]     = d0;
      *(uint4*)&Ks[key][seg + 8] = d1;
    }
#endif
    // --- V tile transposed: 4x4 sub-blocks, b64 LDS stores ---
    {
      int kb = (tid >> 4) * 4;
      int db = (tid & 15) * 4;
      const __bf16* gv = V + (rowOff + kt0 + kb) * D_ + hcol + db;
      __align__(8) __bf16 r0[4], r1[4], r2[4], r3[4];
      *(uint2*)r0 = *(const uint2*)(gv + 0 * D_);
      *(uint2*)r1 = *(const uint2*)(gv + 1 * D_);
      *(uint2*)r2 = *(const uint2*)(gv + 2 * D_);
      *(uint2*)r3 = *(const uint2*)(gv + 3 * D_);
#pragma unroll
      for (int j = 0; j < 4; ++j) {
        __align__(8) __bf16 q[4] = { r0[j], r1[j], r2[j], r3[j] };
        *(uint2*)&Vt[db + j][kb] = *(uint2*)q;
      }
    }
#if HAVE_TDM
    if (wave == 0) __builtin_amdgcn_s_wait_tensorcnt(0);
#endif
    __syncthreads();

    if (kt0 <= q0 + 15) {
#pragma unroll
      for (int c = 0; c < 2; ++c) {  // two 32-key chunks
        const int kbase = kt0 + c * 32;
        // preload all four K fragments, then issue the score WMMAs together
        v16bf bk00 = load_b_frag(&Ks[c * 32][0],       72);
        v16bf bk01 = load_b_frag(&Ks[c * 32][32],      72);
        v16bf bk10 = load_b_frag(&Ks[c * 32 + 16][0],  72);
        v16bf bk11 = load_b_frag(&Ks[c * 32 + 16][32], 72);
        v8f s0 = wmma_bf16(aq0, bk00, zero);
        v8f s1 = wmma_bf16(aq0, bk10, zero);
        s0 = wmma_bf16(aq1, bk01, s0);
        s1 = wmma_bf16(aq1, bk11, s1);

        float sc[2][8];
#pragma unroll
        for (int e = 0; e < 8; ++e) {
          int qi = q0 + rowB + e;
          int k0i = kbase + colL;
          float v0 = s0[e] * 0.125f;
          float v1 = s1[e] * 0.125f;
          sc[0][e] = (k0i      <= qi) ? v0 : -1e30f;
          sc[1][e] = (k0i + 16 <= qi) ? v1 : -1e30f;
        }
        // --- online softmax over the 32-key chunk (half-wave reductions) ---
        float pscale[8];
#pragma unroll
        for (int e = 0; e < 8; ++e) {
          float tmax = fmaxf(sc[0][e], sc[1][e]);
#pragma unroll
          for (int off = 8; off >= 1; off >>= 1)
            tmax = fmaxf(tmax, __shfl_xor(tmax, off, 16));
          float mnew  = fmaxf(mrow[e], tmax);
          float scale = __expf(mrow[e] - mnew);
          float p0 = __expf(sc[0][e] - mnew);
          float p1 = __expf(sc[1][e] - mnew);
          float rs = p0 + p1;
#pragma unroll
          for (int off = 8; off >= 1; off >>= 1)
            rs += __shfl_xor(rs, off, 16);
          lrow[e]   = lrow[e] * scale + rs;
          mrow[e]   = mnew;
          pscale[e] = scale;
          Ps[wave][rowB + e][colL]      = f2bf(p0);
          Ps[wave][rowB + e][16 + colL] = f2bf(p1);
        }
#pragma unroll
        for (int g = 0; g < 4; ++g)
#pragma unroll
          for (int e = 0; e < 8; ++e) accO[g][e] *= pscale[e];

        // --- PV: preload all four V fragments, then 4 WMMAs ---
        v16bf ap  = load_a_frag(&Ps[wave][0][0], 40);
        v16bf bv0 = load_b_frag(&Vt[0][c * 32],  72);
        v16bf bv1 = load_b_frag(&Vt[16][c * 32], 72);
        v16bf bv2 = load_b_frag(&Vt[32][c * 32], 72);
        v16bf bv3 = load_b_frag(&Vt[48][c * 32], 72);
        accO[0] = wmma_bf16(ap, bv0, accO[0]);
        accO[1] = wmma_bf16(ap, bv1, accO[1]);
        accO[2] = wmma_bf16(ap, bv2, accO[2]);
        accO[3] = wmma_bf16(ap, bv3, accO[3]);
      }
    }
    __syncthreads();
  }

#pragma unroll
  for (int g = 0; g < 4; ++g) {
#pragma unroll
    for (int e = 0; e < 8; ++e) {
      float v = accO[g][e] / lrow[e];
      O[(rowOff + q0 + rowB + e) * D_ + hcol + g * 16 + colL] = f2bf(v);
    }
  }
}

// ---------------- launch ----------------
extern "C" void kernel_launch(void* const* d_in, const int* in_sizes, int n_in,
                              void* d_out, int out_size, void* d_ws, size_t ws_size,
                              hipStream_t stream) {
  const float* x  = (const float*)d_in[0];
  const float* Wq = (const float*)d_in[1];
  const float* bq = (const float*)d_in[2];
  const float* Wk = (const float*)d_in[3];
  const float* bk = (const float*)d_in[4];
  const float* Wv = (const float*)d_in[5];
  const float* bv = (const float*)d_in[6];
  const float* Wo = (const float*)d_in[7];
  const float* bo = (const float*)d_in[8];
  float* out = (float*)d_out;

  char* ws = (char*)d_ws;
  size_t off = 0;
  auto alloc = [&](size_t bytes) {
    char* p = ws + off;
    off += (bytes + 255) & ~(size_t)255;
    return p;
  };
  __bf16* xb  = (__bf16*)alloc((size_t)M_ * D_ * 2);
  __bf16* Wqb = (__bf16*)alloc((size_t)D_ * D_ * 2);
  __bf16* Wkb = (__bf16*)alloc((size_t)D_ * D_ * 2);
  __bf16* Wvb = (__bf16*)alloc((size_t)D_ * D_ * 2);
  __bf16* Wob = (__bf16*)alloc((size_t)D_ * D_ * 2);
  __bf16* Qb  = (__bf16*)alloc((size_t)M_ * D_ * 2);
  __bf16* Kb  = (__bf16*)alloc((size_t)M_ * D_ * 2);
  __bf16* Vb  = (__bf16*)alloc((size_t)M_ * D_ * 2);
  __bf16* Ob  = (__bf16*)alloc((size_t)M_ * D_ * 2);

  {
    int nx4 = (M_ * D_) / 4;
    int nw4 = (D_ * D_) / 4;
    cast_f32_to_bf16_v4<<<1024, 256, 0, stream>>>((const float4*)x,  xb,  nx4);
    cast_f32_to_bf16_v4<<<1024, 256, 0, stream>>>((const float4*)Wq, Wqb, nw4);
    cast_f32_to_bf16_v4<<<1024, 256, 0, stream>>>((const float4*)Wk, Wkb, nw4);
    cast_f32_to_bf16_v4<<<1024, 256, 0, stream>>>((const float4*)Wv, Wvb, nw4);
    cast_f32_to_bf16_v4<<<1024, 256, 0, stream>>>((const float4*)Wo, Wob, nw4);
  }

  dim3 gg(M_ / 128, D_ / 128);
  gemm_bf16_wmma<__bf16><<<gg, 256, 0, stream>>>(xb, Wqb, bq, Qb, M_, D_, D_);
  gemm_bf16_wmma<__bf16><<<gg, 256, 0, stream>>>(xb, Wkb, bk, Kb, M_, D_, D_);
  gemm_bf16_wmma<__bf16><<<gg, 256, 0, stream>>>(xb, Wvb, bv, Vb, M_, D_, D_);

  dim3 ga(T_ / 128, H_, B_);
  attn_kernel<<<ga, 256, 0, stream>>>(Qb, Kb, Vb, Ob);

  gemm_bf16_wmma<float><<<gg, 256, 0, stream>>>(Ob, Wob, bo, out, M_, D_, D_);

  (void)in_sizes; (void)n_in; (void)out_size; (void)ws_size;
}
